// TriplaneRenderer_51110110822798
// MI455X (gfx1250) — matrix operand order repaired and checked by reading, not compiled, not run on hardware
//
#include <hip/hip_runtime.h>

typedef __attribute__((ext_vector_type(16))) _Float16 v16h;
typedef __attribute__((ext_vector_type(8)))  float    v8f;

#define LOG2_N 18      // N = 262144 points per batch item
#define CCH    40
#define HW     65536   // 256*256
#define LDA    132     // padded row stride (halves) of the 128x128 activation tile
#define WLD    132     // padded row stride (halves) of the staged weight chunk

// ---- A fragment: 16x32 f16, lane m=l&15, half h=l>>4,
// elements 0..15 hold K = 16h + 0..15 (matches B fragment) ----
__device__ __forceinline__ v16h load_A16(const _Float16* __restrict__ Xs,
                                         int row0, int k0, int lane) {
  const int m = lane & 15;
  const int h = lane >> 4;
  const _Float16* r = Xs + (row0 + m) * LDA + k0 + 16 * h;
  v16h a;
#pragma unroll
  for (int i = 0; i < 8; ++i) a[i] = r[i];
#pragma unroll
  for (int i = 0; i < 8; ++i) a[8 + i] = r[8 + i];
  return a;
}

// ---- B fragment: 32x16 f16, lane n=l&15, lanes 0-15 hold K=0..15,
// lanes 16-31 hold K=16..31. Weights staged as Wb[n][k] so this is one
// contiguous 32-byte LDS read. ----
__device__ __forceinline__ v16h load_B16(const _Float16* __restrict__ Wb,
                                         int nrow0, int k0, int lane) {
  const int n = lane & 15;
  const int h = lane >> 4;
  const _Float16* r = Wb + (nrow0 + n) * WLD + k0 + 16 * h;
  v16h b;
#pragma unroll
  for (int i = 0; i < 16; ++i) b[i] = r[i];
  return b;
}

// Stage a 64-column chunk of a KxHID (row-major, HID=128) f32 weight matrix
// into LDS as f16, transposed to [n][k], zero-padding rows k >= Krows.
__device__ __forceinline__ void stage_weights(_Float16* __restrict__ Wb,
                                              const float* __restrict__ W,
                                              int nbase, int Krows) {
  for (int i = threadIdx.x; i < 64 * 128; i += 256) {
    const int nn = i >> 7;
    const int kk = i & 127;
    const float w = (kk < Krows) ? W[kk * 128 + (nbase + nn)] : 0.0f;
    Wb[nn * WLD + kk] = (_Float16)w;
  }
}

__device__ __forceinline__ float lane16_reduce(float s) {
  s += __shfl_xor(s, 1, 16);
  s += __shfl_xor(s, 2, 16);
  s += __shfl_xor(s, 4, 16);
  s += __shfl_xor(s, 8, 16);
  return s;
}

__global__ __launch_bounds__(256)
void triplane_nerf_wmma(const float* __restrict__ triplane,
                        const float* __restrict__ coords,
                        const float* __restrict__ viewdirs,
                        const float* __restrict__ dw1, const float* __restrict__ db1,
                        const float* __restrict__ dw2, const float* __restrict__ db2,
                        const float* __restrict__ dw3, const float* __restrict__ db3,
                        const float* __restrict__ cw1, const float* __restrict__ cb1,
                        const float* __restrict__ cw2, const float* __restrict__ cb2,
                        const float* __restrict__ cw3, const float* __restrict__ cb3,
                        float* __restrict__ out, int npts)
{
  __shared__ _Float16 buf0[128 * LDA];   // X tile, then reused as hidden tile
  __shared__ _Float16 Wb[64 * WLD];      // staged weight chunk (64 cols)
  __shared__ float    biasS[128];

  const int tid  = threadIdx.x;
  const int lane = tid & 31;
  const int wave = tid >> 5;
  const int base = blockIdx.x * 128;

  // ---------------- Phase 1: bilinear triplane sampling -> f16 tile ----------
  {
    const int p   = tid >> 1;          // local point 0..127
    const int ch0 = (tid & 1) * 20;    // channel half 0..19 / 20..39
    const int g   = base + p;
    const int b   = g >> LOG2_N;
    const float x = coords[g * 3 + 0];
    const float y = coords[g * 3 + 1];
    const float z = coords[g * 3 + 2];
    const float uu[3] = {x, x, y};
    const float vv[3] = {y, z, z};
#pragma unroll
    for (int pl = 0; pl < 3; ++pl) {
      const float u = uu[pl], v = vv[pl];
      const float ixf = ((u + 1.0f) * 256.0f - 1.0f) * 0.5f;
      const float iyf = ((v + 1.0f) * 256.0f - 1.0f) * 0.5f;
      const float fx0 = floorf(ixf), fy0 = floorf(iyf);
      const float fx = ixf - fx0, fy = iyf - fy0;
      const int x0 = (int)fx0, y0 = (int)fy0;
      const int x1 = x0 + 1,  y1 = y0 + 1;
      const float mx0 = ((unsigned)x0 < 256u) ? 1.f : 0.f;
      const float mx1 = ((unsigned)x1 < 256u) ? 1.f : 0.f;
      const float my0 = ((unsigned)y0 < 256u) ? 1.f : 0.f;
      const float my1 = ((unsigned)y1 < 256u) ? 1.f : 0.f;
      const int cx0 = min(max(x0, 0), 255), cx1 = min(max(x1, 0), 255);
      const int cy0 = min(max(y0, 0), 255), cy1 = min(max(y1, 0), 255);
      const int o00 = cy0 * 256 + cx0, o10 = cy0 * 256 + cx1;
      const int o01 = cy1 * 256 + cx0, o11 = cy1 * 256 + cx1;
      const float w00 = (1.f - fx) * (1.f - fy) * mx0 * my0;
      const float w10 = fx * (1.f - fy) * mx1 * my0;
      const float w01 = (1.f - fx) * fy * mx0 * my1;
      const float w11 = fx * fy * mx1 * my1;
      const float* P = triplane + ((size_t)(b * 3 + pl) * CCH + ch0) * HW;
      _Float16* dst = buf0 + p * LDA + pl * 40 + ch0;
      for (int c = 0; c < 20; ++c) {
        const float* pc = P + (size_t)c * HW;
        dst[c] = (_Float16)(pc[o00] * w00 + pc[o10] * w10 +
                            pc[o01] * w01 + pc[o11] * w11);
      }
    }
    if (tid < 128) {  // view dirs at 120..122, zero pad to 128
      const int g2 = base + tid;
      _Float16* dst = buf0 + tid * LDA;
      dst[120] = (_Float16)viewdirs[g2 * 3 + 0];
      dst[121] = (_Float16)viewdirs[g2 * 3 + 1];
      dst[122] = (_Float16)viewdirs[g2 * 3 + 2];
      dst[123] = (_Float16)0.f; dst[124] = (_Float16)0.f;
      dst[125] = (_Float16)0.f; dst[126] = (_Float16)0.f;
      dst[127] = (_Float16)0.f;
    }
  }
  __syncthreads();

  const int row0  = wave * 16;            // this wave's 16-point M tile
  const int ncol  = lane & 15;            // C/D column for this lane
  const int mbase = row0 + (lane >> 4) * 8;  // C/D row base for this lane

  // X fragments stay in VGPRs for both MLP layer-1s; frees buf0 for reuse.
  v16h aX[4];
#pragma unroll
  for (int k = 0; k < 4; ++k) aX[k] = load_A16(buf0, row0, 32 * k, lane);

  // ================= density layer 1: buf0 <- relu(X @ dw1 + db1) ===========
  if (tid < 128) biasS[tid] = db1[tid];
#pragma unroll 1
  for (int half = 0; half < 2; ++half) {
    __syncthreads();
    stage_weights(Wb, dw1, half * 64, 120);   // rows 120..127 zero-padded
    __syncthreads();
    for (int n0 = 0; n0 < 64; n0 += 16) {
      v8f acc = {};
#pragma unroll
      for (int k = 0; k < 4; ++k) {
        v16h bf = load_B16(Wb, n0, 32 * k, lane);
        acc = __builtin_amdgcn_wmma_f32_16x16x32_f16(
            false, aX[k], false, bf, (short)0, acc, false, false);
      }
      const int nc = half * 64 + n0 + ncol;
      const float bb = biasS[nc];
#pragma unroll
      for (int v = 0; v < 8; ++v) {
        float r = acc[v] + bb;
        buf0[(mbase + v) * LDA + nc] = (_Float16)(r > 0.f ? r : 0.f);
      }
    }
  }

  // ======== density layer 2 + fused density head (dw3, db3) ================
  __syncthreads();
  v16h aH[4];
#pragma unroll
  for (int k = 0; k < 4; ++k) aH[k] = load_A16(buf0, row0, 32 * k, lane);
  __syncthreads();
  if (tid < 128) biasS[tid] = db2[tid];

  float dpart[8];
#pragma unroll
  for (int v = 0; v < 8; ++v) dpart[v] = 0.f;

#pragma unroll 1
  for (int half = 0; half < 2; ++half) {
    __syncthreads();
    stage_weights(Wb, dw2, half * 64, 128);
    __syncthreads();
    for (int n0 = 0; n0 < 64; n0 += 16) {
      v8f acc = {};
#pragma unroll
      for (int k = 0; k < 4; ++k) {
        v16h bf = load_B16(Wb, n0, 32 * k, lane);
        acc = __builtin_amdgcn_wmma_f32_16x16x32_f16(
            false, aH[k], false, bf, (short)0, acc, false, false);
      }
      const int nc = half * 64 + n0 + ncol;
      const float bb = biasS[nc];
      const float w3 = dw3[nc];
#pragma unroll
      for (int v = 0; v < 8; ++v) {
        float r = acc[v] + bb;
        r = r > 0.f ? r : 0.f;
        dpart[v] += r * w3;
      }
    }
  }
#pragma unroll
  for (int v = 0; v < 8; ++v) dpart[v] = lane16_reduce(dpart[v]);
  if (ncol == 0) {
    const float b3 = db3[0];
#pragma unroll
    for (int v = 0; v < 8; ++v) out[base + mbase + v] = dpart[v] + b3;
  }

  // ================= color layer 1: buf0 <- relu([X,vd] @ cw1 + cb1) ========
  __syncthreads();
  if (tid < 128) biasS[tid] = cb1[tid];
#pragma unroll 1
  for (int half = 0; half < 2; ++half) {
    __syncthreads();
    stage_weights(Wb, cw1, half * 64, 123);   // rows 123..127 zero-padded
    __syncthreads();
    for (int n0 = 0; n0 < 64; n0 += 16) {
      v8f acc = {};
#pragma unroll
      for (int k = 0; k < 4; ++k) {
        v16h bf = load_B16(Wb, n0, 32 * k, lane);
        acc = __builtin_amdgcn_wmma_f32_16x16x32_f16(
            false, aX[k], false, bf, (short)0, acc, false, false);
      }
      const int nc = half * 64 + n0 + ncol;
      const float bb = biasS[nc];
#pragma unroll
      for (int v = 0; v < 8; ++v) {
        float r = acc[v] + bb;
        buf0[(mbase + v) * LDA + nc] = (_Float16)(r > 0.f ? r : 0.f);
      }
    }
  }

  // ======== color layer 2 + fused rgb head (cw3, cb3, sigmoid) =============
  __syncthreads();
  v16h aG[4];
#pragma unroll
  for (int k = 0; k < 4; ++k) aG[k] = load_A16(buf0, row0, 32 * k, lane);
  __syncthreads();
  if (tid < 128) biasS[tid] = cb2[tid];

  float c0p[8], c1p[8], c2p[8];
#pragma unroll
  for (int v = 0; v < 8; ++v) { c0p[v] = 0.f; c1p[v] = 0.f; c2p[v] = 0.f; }

#pragma unroll 1
  for (int half = 0; half < 2; ++half) {
    __syncthreads();
    stage_weights(Wb, cw2, half * 64, 128);
    __syncthreads();
    for (int n0 = 0; n0 < 64; n0 += 16) {
      v8f acc = {};
#pragma unroll
      for (int k = 0; k < 4; ++k) {
        v16h bf = load_B16(Wb, n0, 32 * k, lane);
        acc = __builtin_amdgcn_wmma_f32_16x16x32_f16(
            false, aG[k], false, bf, (short)0, acc, false, false);
      }
      const int nc = half * 64 + n0 + ncol;
      const float bb = biasS[nc];
      const float w0 = cw3[nc * 3 + 0];
      const float w1 = cw3[nc * 3 + 1];
      const float w2 = cw3[nc * 3 + 2];
#pragma unroll
      for (int v = 0; v < 8; ++v) {
        float r = acc[v] + bb;
        r = r > 0.f ? r : 0.f;
        c0p[v] += r * w0;
        c1p[v] += r * w1;
        c2p[v] += r * w2;
      }
    }
  }
#pragma unroll
  for (int v = 0; v < 8; ++v) {
    c0p[v] = lane16_reduce(c0p[v]);
    c1p[v] = lane16_reduce(c1p[v]);
    c2p[v] = lane16_reduce(c2p[v]);
  }
  if (ncol == 0) {
    const float cb0 = cb3[0], cb1v = cb3[1], cb2v = cb3[2];
#pragma unroll
    for (int v = 0; v < 8; ++v) {
      const int gp = base + mbase + v;
      out[npts + gp * 3 + 0] = 1.f / (1.f + __expf(-(c0p[v] + cb0)));
      out[npts + gp * 3 + 1] = 1.f / (1.f + __expf(-(c1p[v] + cb1v)));
      out[npts + gp * 3 + 2] = 1.f / (1.f + __expf(-(c2p[v] + cb2v)));
    }
  }
}

extern "C" void kernel_launch(void* const* d_in, const int* in_sizes, int n_in,
                              void* d_out, int out_size, void* d_ws, size_t ws_size,
                              hipStream_t stream) {
  (void)n_in; (void)out_size; (void)d_ws; (void)ws_size;
  const float* triplane = (const float*)d_in[0];
  const float* coords   = (const float*)d_in[1];
  const float* viewdirs = (const float*)d_in[2];
  const float* dw1 = (const float*)d_in[3];
  const float* db1 = (const float*)d_in[4];
  const float* dw2 = (const float*)d_in[5];
  const float* db2 = (const float*)d_in[6];
  const float* dw3 = (const float*)d_in[7];
  const float* db3 = (const float*)d_in[8];
  const float* cw1 = (const float*)d_in[9];
  const float* cb1 = (const float*)d_in[10];
  const float* cw2 = (const float*)d_in[11];
  const float* cb2 = (const float*)d_in[12];
  const float* cw3 = (const float*)d_in[13];
  const float* cb3 = (const float*)d_in[14];

  const int npts   = in_sizes[1] / 3;   // B*N = 1,048,576
  const int blocks = npts / 128;        // 8192

  hipLaunchKernelGGL(triplane_nerf_wmma, dim3(blocks), dim3(256), 0, stream,
                     triplane, coords, viewdirs,
                     dw1, db1, dw2, db2, dw3, db3,
                     cw1, cb1, cw2, cb2, cw3, cb3,
                     (float*)d_out, npts);
}
